// GeometryAwareTensorAttention_13176959664206
// MI455X (gfx1250) — matrix operand order repaired and checked by reading, not compiled, not run on hardware
//
#include <hip/hip_runtime.h>
#include <math.h>

// ---------------------------------------------------------------------------
// GeometryAwareTensorAttention for MI455X (gfx1250, wave32, WMMA 16x16x32 f16)
// B=1, N=96, DIM=128, H=8, DH=64, S=5, DI=512, DM=1024
// ---------------------------------------------------------------------------

typedef _Float16 half8 __attribute__((ext_vector_type(8)));
typedef _Float16 v16h  __attribute__((ext_vector_type(16)));
typedef float    v8f   __attribute__((ext_vector_type(8)));

__device__ __forceinline__ v16h cat8(half8 a, half8 b) {
  return __builtin_shufflevector(a, b, 0,1,2,3,4,5,6,7,8,9,10,11,12,13,14,15);
}

// A fragment: 16x32 f16 tile, row-major source (rows m=0..15, stride in halves).
// Lane l<16 : M=l,   K = {k0+0..7, k0+16..23}
// Lane l>=16: M=l-16,K = {k0+8..15, k0+24..31}
__device__ __forceinline__ v16h frag_a(const _Float16* base, int stride, int k0,
                                       int nl, bool lo) {
  const _Float16* rp = base + nl * stride + k0 + (lo ? 0 : 8);
  half8 c0 = *(const half8*)rp;
  half8 c1 = *(const half8*)(rp + 16);
  return cat8(c0, c1);
}

// B fragment: 32x16 f16 tile taken from WT[n][k] (row-major over k, stride Ks).
// Lane l<16 : N=n0+l,    K = {k0+0..7, k0+16..23}
// Lane l>=16: N=n0+l-16, K = {k0+8..15, k0+24..31}
__device__ __forceinline__ v16h frag_b(const _Float16* wt, int stride, int n0,
                                       int k0, int nl, bool lo) {
  const _Float16* rp = wt + (n0 + nl) * stride + k0 + (lo ? 0 : 8);
  half8 c0 = *(const half8*)rp;
  half8 c1 = *(const half8*)(rp + 16);
  return cat8(c0, c1);
}

// fast sigmoid/silu: v_exp_f32 + v_rcp_f32 (error far below the f16 GEMM noise)
__device__ __forceinline__ float sigmoid_fast(float x) {
  return __builtin_amdgcn_rcpf(1.0f + __expf(-x));
}
__device__ __forceinline__ float silu_fast(float x) { return x * sigmoid_fast(x); }

// --------------------------- small prep kernels ----------------------------

// LayerNorm over last dim (128), two gains -> hi, hj. One block per row.
__global__ void ln_kernel(const float* __restrict__ h, const float* __restrict__ g_hi,
                          const float* __restrict__ g_hj, float* __restrict__ hi,
                          float* __restrict__ hj) {
  __shared__ float red[128];
  int i = blockIdx.x, t = threadIdx.x;
  float x = h[i * 128 + t];
  red[t] = x; __syncthreads();
  for (int st = 64; st > 0; st >>= 1) { if (t < st) red[t] += red[t + st]; __syncthreads(); }
  float mu = red[0] * (1.0f / 128.0f);
  __syncthreads();
  float dx = x - mu;
  red[t] = dx * dx; __syncthreads();
  for (int st = 64; st > 0; st >>= 1) { if (t < st) red[t] += red[t + st]; __syncthreads(); }
  float var = red[0] * (1.0f / 128.0f);
  float y = dx * rsqrtf(var + 1e-5f);
  hi[i * 128 + t] = y * g_hi[t];
  hj[i * 128 + t] = y * g_hj[t];
}

// f32 -> f16 copy
__global__ void f2h_kernel(_Float16* __restrict__ dst, const float* __restrict__ src, int n) {
  int id = blockIdx.x * blockDim.x + threadIdx.x;
  if (id < n) dst[id] = (_Float16)src[id];
}

// transpose W[K][N] (f32) -> WT[N][K] (f16)
__global__ void wtrans_kernel(_Float16* __restrict__ dst, const float* __restrict__ src,
                              int K, int N) {
  int id = blockIdx.x * blockDim.x + threadIdx.x;
  if (id >= K * N) return;
  int k = id / N, n = id - k * N;
  dst[n * K + k] = (_Float16)src[id];
}

// q = hi @ Wq, k = hj @ Wk   ([96x128]@[128x512], f32 VALU: exact)
__global__ void proj_qk_kernel(const float* __restrict__ hi, const float* __restrict__ hj,
                               const float* __restrict__ Wq, const float* __restrict__ Wk,
                               float* __restrict__ q, float* __restrict__ kk) {
  int id = blockIdx.x * blockDim.x + threadIdx.x;   // 96*512
  if (id >= 96 * 512) return;
  int i = id >> 9, c = id & 511;
  float aq = 0.f, ak = 0.f;
  const float* hip = hi + i * 128;
  const float* hjp = hj + i * 128;
  for (int k = 0; k < 128; ++k) {
    aq += hip[k] * Wq[k * 512 + c];
    ak += hjp[k] * Wk[k * 512 + c];
  }
  q[id] = aq; kk[id] = ak;
}

// t1 = silu(hj @ W + b)   ([96x128]@[128x1024])
__global__ void mlp1_kernel(const float* __restrict__ hj, const float* __restrict__ W,
                            const float* __restrict__ b, float* __restrict__ t1) {
  int id = blockIdx.x * blockDim.x + threadIdx.x;   // 96*1024
  if (id >= 96 * 1024) return;
  int i = id >> 10, c = id & 1023;
  float acc = b[c];
  const float* hp = hj + i * 128;
  for (int k = 0; k < 128; ++k) acc += hp[k] * W[k * 1024 + c];
  t1[id] = silu_fast(acc);
}

// dst = t1 @ W + b   ([96x1024]@[1024x2560])
__global__ void mlp2_kernel(const float* __restrict__ t1, const float* __restrict__ W,
                            const float* __restrict__ b, float* __restrict__ dst) {
  int id = blockIdx.x * blockDim.x + threadIdx.x;   // 96*2560
  if (id >= 96 * 2560) return;
  int i = id / 2560, c = id - i * 2560;
  float acc = b[c];
  const float* tp = t1 + i * 1024;
  for (int k = 0; k < 1024; ++k) acc += tp[k] * W[k * 2560 + c];
  dst[id] = acc;
}

// gates_raw[i, h*5+s] = sigmoid(hi @ Wg + bg)
__global__ void gates_kernel(const float* __restrict__ hi, const float* __restrict__ Wg,
                             const float* __restrict__ bg, float* __restrict__ gates) {
  int id = blockIdx.x * blockDim.x + threadIdx.x;   // 96*40
  if (id >= 96 * 40) return;
  int i = id / 40, c = id - i * 40;
  float acc = bg[c];
  const float* hp = hi + i * 128;
  for (int k = 0; k < 128; ++k) acc += hp[k] * Wg[k * 40 + c];
  gates[id] = sigmoid_fast(acc);
}

// --------------------------- heavy WMMA kernels ----------------------------

// E1: per 16-row tile of ij rows:  ek = silu(t16 @ Wek)  (WMMA, f32 acc),
// immediately reduced into sim[ij, s*8+h] += ek * q[i,hd] * k[j,hd].
__global__ void e1_kernel(const _Float16* __restrict__ t16, const _Float16* __restrict__ WekT,
                          const float* __restrict__ q, const float* __restrict__ kk,
                          float* __restrict__ sim) {
  __shared__ float sim_s[16 * 40];
  int tid = threadIdx.x;
  int r0 = blockIdx.x * 16;
  for (int idx = tid; idx < 640; idx += 256) sim_s[idx] = 0.f;
  __syncthreads();
  int wave = tid >> 5, lane = tid & 31, nl = lane & 15;
  bool lo = lane < 16;
  const _Float16* abase = t16 + r0 * 128;
  for (int nt = wave; nt < 160; nt += 8) {         // 160 N-tiles of 16 over 2560 cols
    int c0 = nt * 16;
    int s = c0 >> 9, hh = (c0 >> 6) & 7;
    v8f acc = {0.f,0.f,0.f,0.f,0.f,0.f,0.f,0.f};
    #pragma unroll
    for (int ks = 0; ks < 4; ++ks) {               // K = 128
      v16h a = frag_a(abase, 128, ks * 32, nl, lo);
      v16h b = frag_b(WekT, 128, c0, ks * 32, nl, lo);
      acc = __builtin_amdgcn_wmma_f32_16x16x32_f16(false, a, false, b, (short)0,
                                                   acc, false, false);
    }
    int cq = (c0 & 511) + nl;                      // h*64 + d for this lane's column
    #pragma unroll
    for (int r = 0; r < 8; ++r) {
      int m = lo ? r : r + 8;
      int row = r0 + m;
      int i = row / 96, j = row - i * 96;
      float ek = silu_fast(acc[r]);
      float contrib = ek * q[i * 512 + cq] * kk[j * 512 + cq];
      // reduce the 16 n-contributions within each 16-lane half (VALU shuffles),
      // then a single lightly-populated LDS atomic per half.
      #pragma unroll
      for (int off = 1; off < 16; off <<= 1)
        contrib += __shfl_xor(contrib, off, 32);
      if (nl == 0) atomicAdd(&sim_s[m * 40 + s * 8 + hh], contrib);
    }
  }
  __syncthreads();
  for (int idx = tid; idx < 640; idx += 256) {
    int m = idx / 40, c = idx - m * 40;
    sim[(r0 + m) * 40 + c] = sim_s[idx];
  }
}

// softmax over j for each (i, s*8+h); in place.
__global__ void softmax_kernel(float* __restrict__ sim) {
  int id = blockIdx.x * blockDim.x + threadIdx.x;  // 96*40
  if (id >= 96 * 40) return;
  int i = id / 40, c = id - i * 40;
  float* p = sim + i * 96 * 40 + c;
  float mx = -1e30f;
  for (int j = 0; j < 96; ++j) mx = fmaxf(mx, p[j * 40]);
  float sum = 0.f;
  for (int j = 0; j < 96; ++j) sum += expf(p[j * 40] - mx);
  float inv = 1.f / sum;
  for (int j = 0; j < 96; ++j) p[j * 40] = expf(p[j * 40] - mx) * inv;
}

// E2: per 16-row tile, for each s: ev = t16 @ Wev (WMMA) -> LDS,
// seag = gates*(attn*v + ev*pav) -> LDS (f16), out[:,s,:] = seag @ Wo (WMMA).
__global__ void e2_kernel(const _Float16* __restrict__ t16, const _Float16* __restrict__ WevT,
                          const _Float16* __restrict__ WoT, const float* __restrict__ attn,
                          const float* __restrict__ vbuf, const float* __restrict__ pav,
                          const float* __restrict__ gates, float* __restrict__ outb) {
  __shared__ _Float16 seag[16 * 512] __attribute__((aligned(16)));
  int tid = threadIdx.x;
  int r0 = blockIdx.x * 16;
  int wave = tid >> 5, lane = tid & 31, nl = lane & 15;
  bool lo = lane < 16;
  const _Float16* abase = t16 + r0 * 128;
  for (int s = 0; s < 5; ++s) {
    // Phase A: ev tile [16 x 512] for this s
    for (int nt = wave; nt < 32; nt += 8) {
      int c0 = s * 512 + nt * 16;
      v8f acc = {0.f,0.f,0.f,0.f,0.f,0.f,0.f,0.f};
      #pragma unroll
      for (int ks = 0; ks < 4; ++ks) {
        v16h a = frag_a(abase, 128, ks * 32, nl, lo);
        v16h b = frag_b(WevT, 128, c0, ks * 32, nl, lo);
        acc = __builtin_amdgcn_wmma_f32_16x16x32_f16(false, a, false, b, (short)0,
                                                     acc, false, false);
      }
      #pragma unroll
      for (int r = 0; r < 8; ++r) {
        int m = lo ? r : r + 8;
        seag[m * 512 + nt * 16 + nl] = (_Float16)acc[r];
      }
    }
    __syncthreads();
    // Phase B: seag = gates * (attn*v + ev*pav), f16 in LDS
    for (int idx = tid; idx < 16 * 512; idx += 256) {
      int m = idx >> 9, c = idx & 511;
      int row = r0 + m;
      int i = row / 96, j = row - i * 96;
      int hh = c >> 6;
      float ev = (float)seag[idx];
      float av = attn[(i * 96 + j) * 40 + s * 8 + hh];
      float vv = vbuf[j * 2560 + s * 512 + c];
      float pv = pav[j * 2560 + s * 512 + c];
      float g  = gates[i * 40 + hh * 5 + s];
      seag[idx] = (_Float16)(g * (av * vv + ev * pv));
    }
    __syncthreads();
    // Phase C: out[16 x 128] = seag[16 x 512] @ Wo[512 x 128] (one N-tile per wave)
    {
      int c0 = wave * 16;
      v8f acc = {0.f,0.f,0.f,0.f,0.f,0.f,0.f,0.f};
      #pragma unroll
      for (int ks = 0; ks < 16; ++ks) {
        v16h a = frag_a(seag, 512, ks * 32, nl, lo);
        v16h b = frag_b(WoT, 512, c0, ks * 32, nl, lo);
        acc = __builtin_amdgcn_wmma_f32_16x16x32_f16(false, a, false, b, (short)0,
                                                     acc, false, false);
      }
      #pragma unroll
      for (int r = 0; r < 8; ++r) {
        int m = lo ? r : r + 8;
        outb[((r0 + m) * 5 + s) * 128 + c0 + nl] = acc[r];
      }
    }
    __syncthreads();
  }
}

// Final j-contractions: h_out, x1_out, x2_out
__global__ void final_kernel(const float* __restrict__ outb, const float* __restrict__ h,
                             const float* __restrict__ x1, const float* __restrict__ x2,
                             const float* __restrict__ r1, const float* __restrict__ r2,
                             float* __restrict__ dout) {
  int id = blockIdx.x * blockDim.x + threadIdx.x;  // 96*128
  if (id >= 96 * 128) return;
  int i = id >> 7, d = id & 127;
  float ah = h[i * 128 + d];
  float a1[3] = {0.f, 0.f, 0.f};
  float a2[5] = {0.f, 0.f, 0.f, 0.f, 0.f};
  for (int j = 0; j < 96; ++j) {
    const float* o = outb + ((i * 96 + j) * 5) * 128 + d;
    float o0 = o[0], o1 = o[128], o2 = o[256], o3 = o[384], o4 = o[512];
    ah += o0;
    const float* x1p = x1 + (j * 128 + d) * 3;
    const float* r1p = r1 + (i * 96 + j) * 3;
    #pragma unroll
    for (int m = 0; m < 3; ++m) a1[m] += x1p[m] * o3 + r1p[m] * o1;
    const float* x2p = x2 + (j * 128 + d) * 5;
    const float* r2p = r2 + (i * 96 + j) * 5;
    #pragma unroll
    for (int m = 0; m < 5; ++m) a2[m] += x2p[m] * o4 + r2p[m] * o2;
  }
  dout[id] = ah;
  float* x1o = dout + 12288;
  float* x2o = dout + 12288 + 36864;
  #pragma unroll
  for (int m = 0; m < 3; ++m) x1o[id * 3 + m] = a1[m];
  #pragma unroll
  for (int m = 0; m < 5; ++m) x2o[id * 5 + m] = a2[m];
}

// ------------------------------- launcher ----------------------------------

extern "C" void kernel_launch(void* const* d_in, const int* in_sizes, int n_in,
                              void* d_out, int out_size, void* d_ws, size_t ws_size,
                              hipStream_t stream) {
  (void)in_sizes; (void)n_in; (void)out_size; (void)ws_size;
  const float* h    = (const float*)d_in[0];
  const float* t_ij = (const float*)d_in[1];
  const float* r1   = (const float*)d_in[2];
  const float* r2   = (const float*)d_in[3];
  const float* x1   = (const float*)d_in[4];
  const float* x2   = (const float*)d_in[5];
  const float* g_hi = (const float*)d_in[6];
  const float* g_hj = (const float*)d_in[7];
  const float* Wq   = (const float*)d_in[8];
  const float* Wk   = (const float*)d_in[9];
  const float* Wv1  = (const float*)d_in[10];
  const float* bv1  = (const float*)d_in[11];
  const float* Wv2  = (const float*)d_in[12];
  const float* bv2  = (const float*)d_in[13];
  const float* Wek  = (const float*)d_in[14];
  const float* Wev  = (const float*)d_in[15];
  const float* Wp1  = (const float*)d_in[16];
  const float* bp1  = (const float*)d_in[17];
  const float* Wp2  = (const float*)d_in[18];
  const float* bp2  = (const float*)d_in[19];
  const float* Wg   = (const float*)d_in[20];
  const float* bg   = (const float*)d_in[21];
  const float* Wo   = (const float*)d_in[22];
  float* dout = (float*)d_out;

  // workspace carve (f32 region then f16 region); total ~31 MB
  float* fw = (float*)d_ws;
  float* hi_b  = fw; fw += 12288;     // 96*128
  float* hj_b  = fw; fw += 12288;
  float* q_b   = fw; fw += 49152;     // 96*512
  float* k_b   = fw; fw += 49152;
  float* v_b   = fw; fw += 245760;    // 96*2560
  float* p_b   = fw; fw += 245760;
  float* g_b   = fw; fw += 3840;      // 96*40
  float* sim_b = fw; fw += 368640;    // 9216*40
  float* out_b = fw; fw += 5898240;   // 9216*5*128
  float* t1_b  = fw; fw += 98304;     // 96*1024
  _Float16* t16  = (_Float16*)fw;     // 9216*128
  _Float16* WekT = t16 + 1179648;     // 2560*128
  _Float16* WevT = WekT + 327680;
  _Float16* WoT  = WevT + 327680;     // 128*512

  // prep
  ln_kernel<<<96, 128, 0, stream>>>(h, g_hi, g_hj, hi_b, hj_b);
  f2h_kernel<<<4608, 256, 0, stream>>>(t16, t_ij, 1179648);
  wtrans_kernel<<<1280, 256, 0, stream>>>(WekT, Wek, 128, 2560);
  wtrans_kernel<<<1280, 256, 0, stream>>>(WevT, Wev, 128, 2560);
  wtrans_kernel<<<256, 256, 0, stream>>>(WoT, Wo, 512, 128);

  // projections (f32 exact)
  proj_qk_kernel<<<192, 256, 0, stream>>>(hi_b, hj_b, Wq, Wk, q_b, k_b);
  mlp1_kernel<<<384, 256, 0, stream>>>(hj_b, Wv1, bv1, t1_b);
  mlp2_kernel<<<960, 256, 0, stream>>>(t1_b, Wv2, bv2, v_b);
  mlp1_kernel<<<384, 256, 0, stream>>>(hj_b, Wp1, bp1, t1_b);
  mlp2_kernel<<<960, 256, 0, stream>>>(t1_b, Wp2, bp2, p_b);
  gates_kernel<<<15, 256, 0, stream>>>(hi_b, Wg, bg, g_b);

  // heavy WMMA passes
  e1_kernel<<<576, 256, 0, stream>>>(t16, WekT, q_b, k_b, sim_b);
  softmax_kernel<<<15, 256, 0, stream>>>(sim_b);
  e2_kernel<<<576, 256, 0, stream>>>(t16, WevT, WoT, sim_b, v_b, p_b, g_b, out_b);

  // final reductions
  final_kernel<<<48, 256, 0, stream>>>(out_b, h, x1, x2, r1, r2, dout);
}
